// MD5Surrogate_21792664060009
// MI455X (gfx1250) — matrix operand order repaired and checked by reading, not compiled, not run on hardware
//
#include <hip/hip_runtime.h>
#include <hip/hip_bf16.h>
#include <math.h>

typedef _Float16 half_t;
typedef __attribute__((ext_vector_type(16))) _Float16 v16h;
typedef __attribute__((ext_vector_type(8)))  _Float16 v8h;
typedef __attribute__((ext_vector_type(8)))  float    v8f;

#define BATCH    4096
#define D_HID    512
#define N_ROUNDS 64
#define K1PAD    192     // 162 padded to multiple of 32
#define M_TILE   32      // batch rows per workgroup
#define NWAVE    8
#define BLOCK    (NWAVE * 32)
#define XS       200     // xbuf row stride in halfs (padded vs 192)
#define HS       520     // hidden buf row stride in halfs (padded vs 512)

// Branch-free exact-GELU surrogate: erf via Abramowitz-Stegun 7.1.26
// (|abs err| <= 1.5e-7, well below f16 quantization noise).
__device__ __forceinline__ float gelu_fast(float x) {
    float z = fabsf(x) * 0.70710678118654752f;      // |x|/sqrt(2)
    float t = __builtin_amdgcn_rcpf(fmaf(0.3275911f, z, 1.0f));
    float e = __expf(-z * z);
    float p = t * fmaf(t, fmaf(t, fmaf(t, fmaf(t, 1.061405429f, -1.453152027f),
                                       1.421413741f), -0.284496736f), 0.254829592f);
    float er = fmaf(-p, e, 1.0f);                   // erf(|x|/sqrt(2))
    er = copysignf(er, x);
    return 0.5f * x * (1.0f + er);
}

__device__ __forceinline__ int md5_sched(int r) {
    if (r < 16) return r;
    if (r < 32) return (5 * r + 1) & 15;
    if (r < 48) return (3 * r + 5) & 15;
    return (7 * r) & 15;
}

__device__ __forceinline__ int md5_shift(int r) {
    const int base[4][4] = {{7,12,17,22},{5,9,14,20},{4,11,16,23},{6,10,15,21}};
    return base[r >> 4][r & 3];
}

// A-fragment (16x32 f16, MxK) from LDS, per ISA layout:
// lanes 0-15: M=lane, K = kc*32 + {0..7, 16..23}
// lanes 16-31: M=lane-16, K = kc*32 + {8..15, 24..31}
__device__ __forceinline__ v16h load_a(const half_t* buf, int stride, int mtile, int kc, int lane) {
    int row = mtile * 16 + (lane & 15);
    int kb  = kc * 32 + ((lane & 16) ? 8 : 0);
    const half_t* p = buf + row * stride + kb;
    v8h lo = *(const v8h*)p;          // K kb..kb+7
    v8h hi = *(const v8h*)(p + 16);   // K kb+16..kb+23
    return __builtin_shufflevector(lo, hi, 0,1,2,3,4,5,6,7,8,9,10,11,12,13,14,15);
}

// B-fragment (32x16 f16, KxN) for h = x @ W^T: B[k][n] = W[n][k].
// lanes 0-15: N=lane, K = kc*32 + {0..7, 16..23} ; lanes 16-31: N=lane-16, K = kc*32 + {8..15, 24..31}
__device__ __forceinline__ v16h load_b(const half_t* __restrict__ w, int kpad, int ntile, int kc, int lane) {
    int n  = ntile * 16 + (lane & 15);
    int kb = kc * 32 + ((lane & 16) ? 8 : 0);
    const half_t* p = w + (long)n * kpad + kb;
    v8h lo = *(const v8h*)p;
    v8h hi = *(const v8h*)(p + 16);
    return __builtin_shufflevector(lo, hi, 0,1,2,3,4,5,6,7,8,9,10,11,12,13,14,15);
}

__device__ __forceinline__ v8f wmma_f16(v16h a, v16h b, v8f c) {
    return __builtin_amdgcn_wmma_f32_16x16x32_f16(false, a, false, b, (short)0, c, false, false);
}

// 4 adjacent N-tiles sharing each A-fragment, software double-buffered:
// iteration kc+1's A/B loads are issued into fresh registers BEFORE the 4 WMMAs
// of iteration kc, so global latency is covered by in-flight loads instead of
// an s_wait_loadcnt 0 stall per WMMA.
template <int KTILES>
__device__ __forceinline__ void gemm_acc4(v8f acc[4], const half_t* abuf, int astride, int mtile,
                                          const half_t* __restrict__ Wr, int kpad, int nt0, int lane)
{
    v16h a_cur = load_a(abuf, astride, mtile, 0, lane);
    v16h b_cur[4];
    #pragma unroll
    for (int t = 0; t < 4; ++t) b_cur[t] = load_b(Wr, kpad, nt0 + t, 0, lane);

    #pragma unroll
    for (int kc = 0; kc < KTILES; ++kc) {
        v16h a_nxt;
        v16h b_nxt[4];
        if (kc + 1 < KTILES) {
            a_nxt = load_a(abuf, astride, mtile, kc + 1, lane);
            #pragma unroll
            for (int t = 0; t < 4; ++t) b_nxt[t] = load_b(Wr, kpad, nt0 + t, kc + 1, lane);
        }
        #pragma unroll
        for (int t = 0; t < 4; ++t) acc[t] = wmma_f16(a_cur, b_cur[t], acc[t]);
        if (kc + 1 < KTILES) {
            a_cur = a_nxt;
            #pragma unroll
            for (int t = 0; t < 4; ++t) b_cur[t] = b_nxt[t];
        }
    }
}

// 2-tile variant for layer 3.
template <int KTILES>
__device__ __forceinline__ void gemm_acc2(v8f acc[2], const half_t* abuf, int astride, int mtile,
                                          const half_t* __restrict__ Wr, int kpad, int nt0, int lane)
{
    v16h a_cur = load_a(abuf, astride, mtile, 0, lane);
    v16h b_cur[2];
    #pragma unroll
    for (int t = 0; t < 2; ++t) b_cur[t] = load_b(Wr, kpad, nt0 + t, 0, lane);

    #pragma unroll
    for (int kc = 0; kc < KTILES; ++kc) {
        v16h a_nxt;
        v16h b_nxt[2];
        if (kc + 1 < KTILES) {
            a_nxt = load_a(abuf, astride, mtile, kc + 1, lane);
            #pragma unroll
            for (int t = 0; t < 2; ++t) b_nxt[t] = load_b(Wr, kpad, nt0 + t, kc + 1, lane);
        }
        #pragma unroll
        for (int t = 0; t < 2; ++t) acc[t] = wmma_f16(a_cur, b_cur[t], acc[t]);
        if (kc + 1 < KTILES) {
            a_cur = a_nxt;
            #pragma unroll
            for (int t = 0; t < 2; ++t) b_cur[t] = b_nxt[t];
        }
    }
}

// ---- weight conversion kernels (run once per launch into d_ws) ----
__global__ void cvt_pad_w1(const float* __restrict__ src, half_t* __restrict__ dst) {
    const long total = (long)N_ROUNDS * D_HID * K1PAD;
    for (long i = (long)blockIdx.x * blockDim.x + threadIdx.x; i < total;
         i += (long)gridDim.x * blockDim.x) {
        int  k   = (int)(i % K1PAD);
        long row = i / K1PAD;
        dst[i] = (half_t)((k < 162) ? src[row * 162 + k] : 0.0f);
    }
}

__global__ void cvt_f32_f16(const float* __restrict__ src, half_t* __restrict__ dst, long n) {
    for (long i = (long)blockIdx.x * blockDim.x + threadIdx.x; i < n;
         i += (long)gridDim.x * blockDim.x)
        dst[i] = (half_t)src[i];
}

// ---- main fused 64-round MLP kernel ----
__global__ void __launch_bounds__(BLOCK)
md5_mlp_kernel(const float* __restrict__ msg,
               const half_t* __restrict__ W1h, const float* __restrict__ b1,
               const half_t* __restrict__ W2h, const float* __restrict__ b2,
               const half_t* __restrict__ W3h, const float* __restrict__ b3,
               float* __restrict__ out)
{
    __shared__ half_t xbuf [M_TILE * XS];   // 162-wide input (state|word|rinfo|pad), f16
    __shared__ half_t h1buf[M_TILE * HS];   // hidden 1, f16
    __shared__ half_t h2buf[M_TILE * HS];   // hidden 2, f16

    const int tid     = threadIdx.x;
    const int lane    = tid & 31;
    const int wave    = tid >> 5;
    const int rowBase = blockIdx.x * M_TILE;

    // zero xbuf (covers the K padding 162..191 once), then write MD5 init state bits
    for (int i = tid; i < M_TILE * XS; i += BLOCK) xbuf[i] = (half_t)0.0f;
    __syncthreads();
    const unsigned initw[4] = {0x67452301u, 0xefcdab89u, 0x98badcfeu, 0x10325476u};
    for (int i = tid; i < M_TILE * 128; i += BLOCK) {
        int m = i >> 7, k = i & 127;
        xbuf[m * XS + k] = (half_t)(float)((initw[k >> 5] >> (k & 31)) & 1u);
    }
    __syncthreads();

    for (int r = 0; r < N_ROUNDS; ++r) {
        // ---- message word bits (schedule) + round info ----
        const int w  = md5_sched(r);
        const int sh = md5_shift(r);
        for (int i = tid; i < M_TILE * 32; i += BLOCK) {
            int m = i >> 5, j = i & 31;
            float f  = msg[(long)(rowBase + m) * 64 + w * 4 + (j >> 3)];
            int   mb = (int)rintf(f * 255.0f);
            xbuf[m * XS + 128 + j] = (half_t)(float)((mb >> (j & 7)) & 1);
        }
        if (tid < M_TILE * 2) {
            int m = tid >> 1;
            if (tid & 1) xbuf[m * XS + 161] = (half_t)((float)sh / 25.0f);
            else         xbuf[m * XS + 160] = (half_t)((float)r / 64.0f);
        }
        __syncthreads();

        const int mtile = wave >> 2;
        const int nt0   = (wave & 3) * 8;
        const int ncol  = lane & 15;
        const int mb    = mtile * 16 + ((lane & 16) ? 8 : 0);

        // ---- layer 1: (32x192) @ (192x512) -> gelu -> h1 ----
        {
            const half_t* Wr = W1h + (long)r * D_HID * K1PAD;
            const float*  br = b1 + r * D_HID;
            #pragma unroll
            for (int g = 0; g < 2; ++g) {
                const int ntg = nt0 + g * 4;
                float bias[4];
                #pragma unroll
                for (int t = 0; t < 4; ++t) bias[t] = br[(ntg + t) * 16 + ncol];
                v8f acc[4] = {v8f{}, v8f{}, v8f{}, v8f{}};
                gemm_acc4<K1PAD / 32>(acc, xbuf, XS, mtile, Wr, K1PAD, ntg, lane);
                #pragma unroll
                for (int t = 0; t < 4; ++t) {
                    int n = (ntg + t) * 16 + ncol;
                    #pragma unroll
                    for (int i = 0; i < 8; ++i)
                        h1buf[(mb + i) * HS + n] = (half_t)gelu_fast(acc[t][i] + bias[t]);
                }
            }
        }
        __syncthreads();

        // ---- layer 2: (32x512) @ (512x512) -> gelu -> h2 ----
        {
            const half_t* Wr = W2h + (long)r * D_HID * D_HID;
            const float*  br = b2 + r * D_HID;
            #pragma unroll
            for (int g = 0; g < 2; ++g) {
                const int ntg = nt0 + g * 4;
                float bias[4];
                #pragma unroll
                for (int t = 0; t < 4; ++t) bias[t] = br[(ntg + t) * 16 + ncol];
                v8f acc[4] = {v8f{}, v8f{}, v8f{}, v8f{}};
                gemm_acc4<D_HID / 32>(acc, h1buf, HS, mtile, Wr, D_HID, ntg, lane);
                #pragma unroll
                for (int t = 0; t < 4; ++t) {
                    int n = (ntg + t) * 16 + ncol;
                    #pragma unroll
                    for (int i = 0; i < 8; ++i)
                        h2buf[(mb + i) * HS + n] = (half_t)gelu_fast(acc[t][i] + bias[t]);
                }
            }
        }
        __syncthreads();

        // ---- layer 3: (32x512) @ (512x128) -> new state (f16 into xbuf; f32 out on last round) ----
        {
            const half_t* Wr = W3h + (long)r * 128 * D_HID;
            const float*  br = b3 + r * 128;
            const int ntA = (wave * 2) & 7;
            float bias[2];
            bias[0] = br[ntA * 16 + ncol];
            bias[1] = br[(ntA + 1) * 16 + ncol];
            v8f acc[2] = {v8f{}, v8f{}};
            gemm_acc2<D_HID / 32>(acc, h2buf, HS, mtile, Wr, D_HID, ntA, lane);
            #pragma unroll
            for (int t = 0; t < 2; ++t) {
                int n = (ntA + t) * 16 + ncol;
                #pragma unroll
                for (int i = 0; i < 8; ++i) {
                    float v = acc[t][i] + bias[t];
                    xbuf[(mb + i) * XS + n] = (half_t)v;
                    if (r == N_ROUNDS - 1)
                        out[(long)(rowBase + mb + i) * 128 + n] = v;
                }
            }
        }
        __syncthreads();
    }
}

extern "C" void kernel_launch(void* const* d_in, const int* in_sizes, int n_in,
                              void* d_out, int out_size, void* d_ws, size_t ws_size,
                              hipStream_t stream) {
    const float* msg = (const float*)d_in[0];
    const float* W1  = (const float*)d_in[1];
    const float* b1  = (const float*)d_in[2];
    const float* W2  = (const float*)d_in[3];
    const float* b2  = (const float*)d_in[4];
    const float* W3  = (const float*)d_in[5];
    const float* b3  = (const float*)d_in[6];
    float* out = (float*)d_out;

    // f16 weight images in workspace: W1 padded [64][512][192], W2 [64][512][512], W3 [64][128][512]
    half_t* W1h = (half_t*)d_ws;
    half_t* W2h = W1h + (long)N_ROUNDS * D_HID * K1PAD;
    half_t* W3h = W2h + (long)N_ROUNDS * D_HID * D_HID;

    cvt_pad_w1<<<2048, 256, 0, stream>>>(W1, W1h);
    cvt_f32_f16<<<2048, 256, 0, stream>>>(W2, W2h, (long)N_ROUNDS * D_HID * D_HID);
    cvt_f32_f16<<<1024, 256, 0, stream>>>(W3, W3h, (long)N_ROUNDS * 128 * D_HID);

    md5_mlp_kernel<<<BATCH / M_TILE, BLOCK, 0, stream>>>(msg, W1h, b1, W2h, b2, W3h, b3, out);
}
